// EnergyMHA_30614526886141
// MI455X (gfx1250) — compile-verified, matrix operand order checked by reading, not run on hardware
//
#include <hip/hip_runtime.h>
#include <hip/hip_bf16.h>

// ---------------------------------------------------------------------------
// EnergyMHA forward energy + grad wrt x, CDNA5 (gfx1250) WMMA implementation.
// B=2, L=2048, D=1024, H=16, Z=64. All GEMMs on v_wmma_f32_16x16x32_bf16.
// All operands (including transposed copies) have K contiguous so every
// WMMA operand is fed by two global_load_b128 per lane per K-step.
// ---------------------------------------------------------------------------

#define BB 2
#define LL 2048
#define DD 1024
#define HH 16
#define ZZ 64
#define HZ (HH * ZZ)      // 1024
#define BL (BB * LL)      // 4096

typedef __attribute__((ext_vector_type(16))) __bf16 bf16x16;
typedef __attribute__((ext_vector_type(8)))  __bf16 bf16x8;
typedef __attribute__((ext_vector_type(8)))  float  f32x8;

__device__ __forceinline__ f32x8 wmma_bf16(bf16x16 a, bf16x16 b, f32x8 c) {
  // D = A(16x32 bf16) * B(32x16 bf16) + C(16x16 f32)
  return __builtin_amdgcn_wmma_f32_16x16x32_bf16(
      /*neg_a=*/false, a, /*neg_b=*/false, b,
      /*c_mod=*/(short)0, c, /*reuse_a=*/false, /*reuse_b=*/false);
}

// Load a 32-wide K slice for one row/column when K is contiguous in memory:
// two 16-byte chunks at element offsets {half8, 16+half8}, half8 = 8*(lane>=16)
// (matches the ISA 16-bit A/B operand layout for 16x16x32).
__device__ __forceinline__ bf16x16 load_row16(const __bf16* base, int lane) {
  const int h8 = (lane & 16) >> 1;  // 0 or 8
  bf16x8 lo = *(const bf16x8*)(base + h8);
  bf16x8 hi = *(const bf16x8*)(base + 16 + h8);
  return __builtin_shufflevector(lo, hi, 0, 1, 2, 3, 4, 5, 6, 7,
                                 8, 9, 10, 11, 12, 13, 14, 15);
}

// ---------------------------------------------------------------------------
// K0a: fp32 -> bf16 conversion (x)
// ---------------------------------------------------------------------------
__global__ void emha_cvt_kernel(const float* __restrict__ in,
                                __bf16* __restrict__ out, int n) {
  int i = blockIdx.x * 256 + threadIdx.x;
  if (i < n) out[i] = (__bf16)in[i];
}

// ---------------------------------------------------------------------------
// K0b: fp32 -> bf16 convert + transpose (W): outN (rows,cols), outT (cols,rows)
// ---------------------------------------------------------------------------
__global__ void emha_cvtT_kernel(const float* __restrict__ in,
                                 __bf16* __restrict__ outN,
                                 __bf16* __restrict__ outT, int rows,
                                 int cols) {
  int i = blockIdx.x * 256 + threadIdx.x;
  if (i < rows * cols) {
    int r = i / cols, c = i - r * cols;
    float v = in[i];
    outN[i] = (__bf16)v;
    outT[(size_t)c * rows + r] = (__bf16)v;
  }
}

// ---------------------------------------------------------------------------
// K1: projection GEMM  O[m,n] = sum_k X[m,k] * W[n,k]   (M=4096,N=1024,K=1024)
// Writes normal (b,l,hz) layout and transposed (b,hz,l) layout (16B vector
// store per lane: the 8 accumulator rows are contiguous along l).
// ---------------------------------------------------------------------------
__global__ void emha_proj_kernel(const __bf16* __restrict__ X,
                                 const __bf16* __restrict__ W,
                                 __bf16* __restrict__ O,
                                 __bf16* __restrict__ OT) {
  const int lane = threadIdx.x & 31;
  const int wave = threadIdx.x >> 5;
  const int n0 = (blockIdx.x * 4 + wave) * 16;
  const int m0 = blockIdx.y * 16;
  const int ln = lane & 15, half = lane >> 4;
  const __bf16* arow = X + (size_t)(m0 + ln) * DD;  // A row (K contiguous)
  const __bf16* brow = W + (size_t)(n0 + ln) * DD;  // B col = W row (K contig)
  f32x8 acc = {};
  for (int k = 0; k < DD; k += 32) {
    bf16x16 a = load_row16(arow + k, lane);
    bf16x16 b = load_row16(brow + k, lane);
    acc = wmma_bf16(a, b, acc);
  }
  const int mrow = m0 + 8 * half;
  bf16x8 tvec;
#pragma unroll
  for (int r = 0; r < 8; ++r) {
    __bf16 v = (__bf16)acc[r];
    O[(size_t)(mrow + r) * HZ + n0 + ln] = v;  // coalesced across lanes
    tvec[r] = v;
  }
  const int b = m0 >> 11;            // which batch (L=2048 rows per batch)
  const int l0 = (m0 & (LL - 1)) + 8 * half;
  *(bf16x8*)(OT + (size_t)(b * HZ + n0 + ln) * LL + l0) = tvec;
}

// ---------------------------------------------------------------------------
// K2: zero the two energy accumulators
// ---------------------------------------------------------------------------
__global__ void emha_init_kernel(float* __restrict__ en) {
  if (threadIdx.x < BB) en[threadIdx.x] = 0.0f;
}

// ---------------------------------------------------------------------------
// K3: flash pass 1 — logsumexp over k of beta*att[q,k], transposed iteration:
// S'[k,q] = sum_z qp[k,z]*kp[q,z].  Wave owns 16 q COLUMNS (B operand fixed,
// loaded once); k runs over M-tiles.  The softmax reduction is then over the
// 8 accumulator regs within each lane + sequential tiles: no shuffles in the
// hot loop, one shfl_xor(16) merge at the end.  Also accumulates
// en[b] += (-1/beta) * sum lse.
// ---------------------------------------------------------------------------
__global__ void emha_lse_kernel(const __bf16* __restrict__ kp,
                                const __bf16* __restrict__ qp,
                                const float* __restrict__ beta,
                                float* __restrict__ lse,
                                float* __restrict__ en) {
  const int lane = threadIdx.x & 31;
  const int q0 = blockIdx.x * 16;
  const int h = blockIdx.y;
  const int b = blockIdx.z;
  const int ln = lane & 15;
  const float b0 = beta[0];

  // B operand: columns q (kp rows), fixed for the whole loop.
  const __bf16* bbase = kp + ((size_t)(b * LL + q0 + ln) * HZ + h * ZZ);
  bf16x16 bv0 = load_row16(bbase, lane);       // z = 0..31
  bf16x16 bv1 = load_row16(bbase + 32, lane);  // z = 32..63

  float run_m = -INFINITY, run_s = 0.0f;
  for (int kt = 0; kt < LL; kt += 16) {
    const __bf16* abase = qp + ((size_t)(b * LL + kt + ln) * HZ + h * ZZ);
    bf16x16 a0 = load_row16(abase, lane);
    bf16x16 a1 = load_row16(abase + 32, lane);
    f32x8 c = {};
    c = wmma_bf16(a0, bv0, c);
    c = wmma_bf16(a1, bv1, c);
    float tm = b0 * c[0];
#pragma unroll
    for (int r = 1; r < 8; ++r) tm = fmaxf(tm, b0 * c[r]);
    float nm = fmaxf(run_m, tm);
    float acc = 0.0f;
#pragma unroll
    for (int r = 0; r < 8; ++r) acc += __expf(b0 * c[r] - nm);
    run_s = run_s * __expf(run_m - nm) + acc;
    run_m = nm;
  }
  // Merge the two half-wave row groups (k = .. and k+8 ..).
  float om = __shfl_xor(run_m, 16);
  float os = __shfl_xor(run_s, 16);
  float nm = fmaxf(run_m, om);
  float s = run_s * __expf(run_m - nm) + os * __expf(om - nm);
  float l = nm + __logf(s);
  if (lane < 16) lse[(size_t)(b * HH + h) * LL + q0 + ln] = l;  // coalesced
  float ps = l;
  ps += __shfl_xor(ps, 1);
  ps += __shfl_xor(ps, 2);
  ps += __shfl_xor(ps, 4);
  ps += __shfl_xor(ps, 8);
  if (lane == 0) atomicAdd(&en[b], (-1.0f / b0) * ps);
}

// ---------------------------------------------------------------------------
// K4: flash pass 2 — recompute S tiles, P = -exp(beta*S - lse), and
//   mode 0: Gk[q,z] = sum_k P[q,k] * qp[k,z]   (lse indexed by row)
//   mode 1: Gq[k,z] = sum_q P'[k,q] * kp[q,z]  (S' = qp.kp^T, lse by column)
// P tile transposed C-layout -> A-layout through LDS (1 KB / wave).
// Second-stage B operand comes from the (b,hz,l) transposed copy so K (=l)
// is contiguous (b128 loads instead of strided u16 gathers).
// ---------------------------------------------------------------------------
__global__ void emha_grad_kernel(const __bf16* __restrict__ kp,
                                 const __bf16* __restrict__ qp,
                                 const __bf16* __restrict__ kpT,
                                 const __bf16* __restrict__ qpT,
                                 const float* __restrict__ lse,
                                 const float* __restrict__ beta,
                                 __bf16* __restrict__ Gk,
                                 __bf16* __restrict__ Gq) {
  __shared__ __align__(16) __bf16 psh[16 * 32];
  const int lane = threadIdx.x & 31;
  const int m0 = blockIdx.x * 16;
  const int h = blockIdx.y;
  const int b = blockIdx.z >> 1;
  const int mode = blockIdx.z & 1;
  const int half = lane >> 4, ln = lane & 15;
  const float b0 = beta[0];

  const __bf16* own = mode ? qp : kp;
  const __bf16* oth = mode ? kp : qp;
  const __bf16* othT = mode ? kpT : qpT;
  __bf16* G = mode ? Gq : Gk;
  const float* lrow = lse + (size_t)(b * HH + h) * LL;

  const __bf16* abase = own + ((size_t)(b * LL + m0 + ln) * HZ + h * ZZ);
  bf16x16 a0 = load_row16(abase, lane);
  bf16x16 a1 = load_row16(abase + 32, lane);

  float lse_m[8];
  if (mode == 0) {
#pragma unroll
    for (int r = 0; r < 8; ++r) lse_m[r] = lrow[m0 + 8 * half + r];
  }

  f32x8 zero = {};
  f32x8 g[4] = {zero, zero, zero, zero};

  for (int ot = 0; ot < LL; ot += 32) {
#pragma unroll
    for (int sub = 0; sub < 2; ++sub) {
      const int n0 = ot + sub * 16;
      const __bf16* bbase = oth + ((size_t)(b * LL + n0 + ln) * HZ + h * ZZ);
      bf16x16 bv0 = load_row16(bbase, lane);
      bf16x16 bv1 = load_row16(bbase + 32, lane);
      f32x8 c = {};
      c = wmma_bf16(a0, bv0, c);
      c = wmma_bf16(a1, bv1, c);
      const float lc = mode ? lrow[n0 + ln] : 0.0f;
#pragma unroll
      for (int r = 0; r < 8; ++r) {
        float v = b0 * c[r];
        float l = mode ? lc : lse_m[r];
        float p = -__expf(v - l);  // minus sign of the gradient folded in
        psh[(8 * half + r) * 32 + sub * 16 + ln] = (__bf16)p;
      }
    }
    __syncthreads();
    bf16x16 pa = load_row16(psh + ln * 32, lane);  // A-layout: row=ln, K=kmap
#pragma unroll
    for (int zt = 0; zt < 4; ++zt) {
      // B[kk][z]: column z = zt*16+ln, K = l index contiguous in othT.
      const __bf16* b2 =
          othT + (size_t)(b * HZ + h * ZZ + zt * 16 + ln) * LL + ot;
      bf16x16 bv = load_row16(b2, lane);
      g[zt] = wmma_bf16(pa, bv, g[zt]);
    }
    __syncthreads();
  }
#pragma unroll
  for (int zt = 0; zt < 4; ++zt)
#pragma unroll
    for (int r = 0; r < 8; ++r)
      G[(size_t)(b * LL + m0 + 8 * half + r) * HZ + h * ZZ + zt * 16 + ln] =
          (__bf16)g[zt][r];
}

// ---------------------------------------------------------------------------
// K5: gx[m,d] = sum_hz Gk[m,hz]*Wk[hz,d] + Gq[m,hz]*Wq[hz,d]   (fp32 out)
// W operands from transposed (D,HZ) copies so K (=hz) is contiguous.
// ---------------------------------------------------------------------------
__global__ void emha_gx_kernel(const __bf16* __restrict__ Gk,
                               const __bf16* __restrict__ Gq,
                               const __bf16* __restrict__ WkT,
                               const __bf16* __restrict__ WqT,
                               float* __restrict__ gx) {
  const int lane = threadIdx.x & 31;
  const int wave = threadIdx.x >> 5;
  const int n0 = (blockIdx.x * 4 + wave) * 16;
  const int m0 = blockIdx.y * 16;
  const int ln = lane & 15;
  const __bf16* ak = Gk + (size_t)(m0 + ln) * HZ;
  const __bf16* aq = Gq + (size_t)(m0 + ln) * HZ;
  const __bf16* bk = WkT + (size_t)(n0 + ln) * HZ;  // column d, K=hz contig
  const __bf16* bq = WqT + (size_t)(n0 + ln) * HZ;
  f32x8 acc = {};
  for (int k = 0; k < HZ; k += 32) {
    bf16x16 a1 = load_row16(ak + k, lane);
    bf16x16 b1 = load_row16(bk + k, lane);
    acc = wmma_bf16(a1, b1, acc);
    bf16x16 a2 = load_row16(aq + k, lane);
    bf16x16 b2 = load_row16(bq + k, lane);
    acc = wmma_bf16(a2, b2, acc);
  }
  const int mrow = m0 + 8 * (lane >> 4);
#pragma unroll
  for (int r = 0; r < 8; ++r)
    gx[(size_t)(mrow + r) * DD + n0 + ln] = acc[r];
}

// ---------------------------------------------------------------------------
// Host-side launch
// ---------------------------------------------------------------------------
extern "C" void kernel_launch(void* const* d_in, const int* in_sizes, int n_in,
                              void* d_out, int out_size, void* d_ws,
                              size_t ws_size, hipStream_t stream) {
  const float* x = (const float*)d_in[0];     // (B,L,D)
  const float* Wq = (const float*)d_in[1];    // (H,Z,D)
  const float* Wk = (const float*)d_in[2];    // (H,Z,D)
  const float* beta = (const float*)d_in[3];  // (1,)

  float* out = (float*)d_out;
  float* en = out;       // (B,)
  float* gx = out + BB;  // (B,L,D)

  // Workspace layout (all 16B aligned)
  char* w = (char*)d_ws;
  __bf16* xh = (__bf16*)w;  w += (size_t)BL * DD * sizeof(__bf16);    // 8 MB
  __bf16* wqh = (__bf16*)w; w += (size_t)HZ * DD * sizeof(__bf16);    // 2 MB
  __bf16* wkh = (__bf16*)w; w += (size_t)HZ * DD * sizeof(__bf16);    // 2 MB
  __bf16* wqt = (__bf16*)w; w += (size_t)HZ * DD * sizeof(__bf16);    // 2 MB
  __bf16* wkt = (__bf16*)w; w += (size_t)HZ * DD * sizeof(__bf16);    // 2 MB
  __bf16* qp = (__bf16*)w;  w += (size_t)BL * HZ * sizeof(__bf16);    // 8 MB
  __bf16* kp = (__bf16*)w;  w += (size_t)BL * HZ * sizeof(__bf16);    // 8 MB
  __bf16* qpT = (__bf16*)w; w += (size_t)BL * HZ * sizeof(__bf16);    // 8 MB
  __bf16* kpT = (__bf16*)w; w += (size_t)BL * HZ * sizeof(__bf16);    // 8 MB
  float* lse = (float*)w;   w += (size_t)BB * HH * LL * sizeof(float);// 256 KB
  __bf16* Gk = (__bf16*)w;  w += (size_t)BL * HZ * sizeof(__bf16);    // 8 MB
  __bf16* Gq = (__bf16*)w;  w += (size_t)BL * HZ * sizeof(__bf16);    // 8 MB

  // K0: fp32 -> bf16 operand copies (+ transposed W copies)
  const int NX = BL * DD, NW = HZ * DD;
  emha_cvt_kernel<<<(NX + 255) / 256, 256, 0, stream>>>(x, xh, NX);
  emha_cvtT_kernel<<<(NW + 255) / 256, 256, 0, stream>>>(Wq, wqh, wqt, HZ, DD);
  emha_cvtT_kernel<<<(NW + 255) / 256, 256, 0, stream>>>(Wk, wkh, wkt, HZ, DD);

  // K1: projections qp = x Wq^T, kp = x Wk^T (normal + transposed layouts)
  dim3 pgrid(HZ / 64, BL / 16), pblk(128);
  emha_proj_kernel<<<pgrid, pblk, 0, stream>>>(xh, wqh, qp, qpT);
  emha_proj_kernel<<<pgrid, pblk, 0, stream>>>(xh, wkh, kp, kpT);

  // K2 + K3: lse and energy
  emha_init_kernel<<<1, 32, 0, stream>>>(en);
  emha_lse_kernel<<<dim3(LL / 16, HH, BB), 32, 0, stream>>>(kp, qp, beta, lse,
                                                            en);

  // K4: Gk (mode 0) and Gq (mode 1)
  emha_grad_kernel<<<dim3(LL / 16, HH, 2 * BB), 32, 0, stream>>>(
      kp, qp, kpT, qpT, lse, beta, Gk, Gq);

  // K5: gx = Gk Wk + Gq Wq
  emha_gx_kernel<<<dim3(DD / 64, BL / 16), 128, 0, stream>>>(Gk, Gq, wkt, wqt,
                                                             gx);
}
